// FusionModel_76811195122110
// MI455X (gfx1250) — matrix operand order repaired and checked by reading, not compiled, not run on hardware
//
#include <hip/hip_runtime.h>
#include <math.h>

// ---------------- model dims (fixed by the reference) ----------------
#define SV   3072
#define SA   768
#define CTXN 512
#define DM   1536
#define NH   12
#define HDM  128
#define FF   6144
#define EPSF 1e-6f
#define ATT_SCALE 0.08838834764831845f   // 1/sqrt(128)

typedef __attribute__((ext_vector_type(16))) __bf16 v16bf;
typedef __attribute__((ext_vector_type(8)))  __bf16 v8bf;
typedef __attribute__((ext_vector_type(8)))  float  v8f;
typedef __attribute__((ext_vector_type(4)))  int    v4i;

// CDNA5 async global->LDS copy path (ASYNCcnt), guarded so either toolchain compiles.
#if defined(__gfx1250__) && __has_builtin(__builtin_amdgcn_global_load_async_to_lds_b128) && \
    __has_builtin(__builtin_amdgcn_s_wait_asynccnt)
#define HAVE_ASYNC_LDS 1
#else
#define HAVE_ASYNC_LDS 0
#endif

__device__ __forceinline__ float gelu_tanh(float x) {
  float x3 = x * x * x;
  return 0.5f * x * (1.0f + tanhf(0.7978845608028654f * (x + 0.044715f * x3)));
}

// ---------------- WMMA GEMM: C[M][N] = epi(A_f32[M][K] x Bt_bf16[N][K]^T + bias[N]) ----------------
// Block tile 128x128, 8 wave32 waves in 4(M)x2(N) grid; wave tile 32x64 = 2x4 WMMA frags.
// Requires M%128==0, N%128==0, K%32==0 (true for every GEMM in this model).
#define BM   128
#define BN   128
#define KSTP 32
#define ASTR 40   // LDS row stride (elems); 80B = 5*16B -> 16B-aligned vector access
#define BSTR 40

template <int EPI>
__global__ __launch_bounds__(256)
void gemm_bf16_kernel(const float* __restrict__ A, int lda,
                      const __bf16* __restrict__ Bt, int ldb,
                      const float* __restrict__ bias,
                      float* __restrict__ C, int ldc, int K) {
  __shared__ __attribute__((aligned(16))) __bf16 As[BM * ASTR];
  __shared__ __attribute__((aligned(16))) __bf16 Bs[BN * BSTR];

  const int tid  = threadIdx.x;
  const int m0   = blockIdx.y * BM;
  const int n0   = blockIdx.x * BN;
  const int wave = tid >> 5;
  const int lane = tid & 31;
  const int wm   = (wave & 3) * 32;    // wave row offset (4 waves over M)
  const int wn   = (wave >> 2) * 64;   // wave col offset (2 waves over N)
  const int half = lane >> 4;
  const int l16  = lane & 15;

  v8f acc[2][4];
  for (int i = 0; i < 2; ++i)
    for (int j = 0; j < 4; ++j)
      for (int r = 0; r < 8; ++r) acc[i][j][r] = 0.0f;

  const int ar = tid >> 3;        // 0..31  (A tile row group)
  const int ac = (tid & 7) * 4;   // 0..28  (A tile col, float4)
  const int br = tid >> 2;        // 0..63  (B tile row = N index; +64 for second half)
  const int bc = (tid & 3) * 8;   // 0..24  (B tile col, 8 bf16)

  for (int k0 = 0; k0 < K; k0 += KSTP) {
    // stage B tile: 128(N) x 32(K) bf16; rows of Bt are contiguous in K.
#if HAVE_ASYNC_LDS
    __builtin_amdgcn_global_load_async_to_lds_b128(
        (__attribute__((address_space(1))) v4i*)(Bt + (size_t)(n0 + br) * ldb + (k0 + bc)),
        (__attribute__((address_space(3))) v4i*)(&Bs[br * BSTR + bc]), 0, 0);
    __builtin_amdgcn_global_load_async_to_lds_b128(
        (__attribute__((address_space(1))) v4i*)(Bt + (size_t)(n0 + br + 64) * ldb + (k0 + bc)),
        (__attribute__((address_space(3))) v4i*)(&Bs[(br + 64) * BSTR + bc]), 0, 0);
#else
    {
      const v8bf v0 = *reinterpret_cast<const v8bf*>(Bt + (size_t)(n0 + br) * ldb + (k0 + bc));
      *reinterpret_cast<v8bf*>(&Bs[br * BSTR + bc]) = v0;
      const v8bf v1 = *reinterpret_cast<const v8bf*>(Bt + (size_t)(n0 + br + 64) * ldb + (k0 + bc));
      *reinterpret_cast<v8bf*>(&Bs[(br + 64) * BSTR + bc]) = v1;
    }
#endif
    // stage A tile: 128x32 f32 -> bf16 LDS (conversion must go through VGPRs)
#pragma unroll
    for (int i = 0; i < 4; ++i) {
      const int rr = ar + 32 * i;
      const float4 f = *reinterpret_cast<const float4*>(A + (size_t)(m0 + rr) * lda + (k0 + ac));
      union { __bf16 h[4]; uint2 u; } pk;
      pk.h[0] = (__bf16)f.x; pk.h[1] = (__bf16)f.y; pk.h[2] = (__bf16)f.z; pk.h[3] = (__bf16)f.w;
      *reinterpret_cast<uint2*>(&As[rr * ASTR + ac]) = pk.u;
    }
#if HAVE_ASYNC_LDS
    __builtin_amdgcn_s_wait_asynccnt(0);
#endif
    __syncthreads();

    // A fragment: lane m=l16, elems e<8 -> K=half*8+e ; e>=8 -> K=16+half*8+(e-8)
    v16bf afr[2], bfr[4];
#pragma unroll
    for (int i = 0; i < 2; ++i) {
      const __bf16* rb = &As[(wm + i * 16 + l16) * ASTR + half * 8];
      v8bf lo = *reinterpret_cast<const v8bf*>(rb);
      v8bf hi = *reinterpret_cast<const v8bf*>(rb + 16);
      afr[i] = __builtin_shufflevector(lo, hi, 0,1,2,3,4,5,6,7,8,9,10,11,12,13,14,15);
    }
    // B fragment: lane n=l16, elems e<8 -> K=half*8+e ; e>=8 -> K=16+half*8+(e-8)  (Bs is N-major, K contiguous)
#pragma unroll
    for (int j = 0; j < 4; ++j) {
      const __bf16* rb = &Bs[(wn + j * 16 + l16) * BSTR + half * 8];
      v8bf lo = *reinterpret_cast<const v8bf*>(rb);
      v8bf hi = *reinterpret_cast<const v8bf*>(rb + 16);
      bfr[j] = __builtin_shufflevector(lo, hi, 0,1,2,3,4,5,6,7,8,9,10,11,12,13,14,15);
    }
#pragma unroll
    for (int i = 0; i < 2; ++i)
#pragma unroll
      for (int j = 0; j < 4; ++j)
        acc[i][j] = __builtin_amdgcn_wmma_f32_16x16x32_bf16(
            false, afr[i], false, bfr[j], (short)0, acc[i][j], false, false);
    __syncthreads();
  }

  // epilogue: C VGPR r holds M = half*8 + r, N = l16
#pragma unroll
  for (int j = 0; j < 4; ++j) {
    const int col = n0 + wn + j * 16 + l16;
    const float bv = bias ? bias[col] : 0.0f;
#pragma unroll
    for (int i = 0; i < 2; ++i) {
      const int rbase = m0 + wm + i * 16 + half * 8;
#pragma unroll
      for (int r = 0; r < 8; ++r) {
        float v = acc[i][j][r] + bv;
        if (EPI == 1) v = gelu_tanh(v);
        C[(size_t)(rbase + r) * ldc + col] = v;
      }
    }
  }
}

// ---------------- transpose + convert: in f32 [R][C] -> out bf16 [C][R] ----------------
__global__ __launch_bounds__(256)
void transpose_conv_kernel(const float* __restrict__ in, __bf16* __restrict__ out, int R, int Cc) {
  __shared__ float tile[32][33];
  const int c0 = blockIdx.x * 32;
  const int r0 = blockIdx.y * 32;
  const int tx = threadIdx.x;   // 0..31
  const int ty = threadIdx.y;   // 0..7
#pragma unroll
  for (int i = 0; i < 32; i += 8)
    tile[ty + i][tx] = in[(size_t)(r0 + ty + i) * Cc + (c0 + tx)];
  __syncthreads();
#pragma unroll
  for (int i = 0; i < 32; i += 8)
    out[(size_t)(c0 + ty + i) * R + (r0 + tx)] = (__bf16)tile[tx][ty + i];
}

// ---------------- elementwise convert f32 -> bf16 ----------------
__global__ void conv_bf16_kernel(const float* __restrict__ in, __bf16* __restrict__ out, size_t n) {
  size_t i = (size_t)blockIdx.x * 256 + threadIdx.x;
  if (i < n) out[i] = (__bf16)in[i];
}

// ---------------- LayerNorm (+optional affine, +optional AdaLN modulation from e6) ----------------
// out = ln(x) [*g + b] ; if e6: out = out*(1+e6[s,isc,:]) + e6[s,ish,:]
__global__ __launch_bounds__(256)
void ln_kernel(const float* __restrict__ x, const float* __restrict__ g, const float* __restrict__ b,
               const float* __restrict__ e6, int ish, int isc, float* __restrict__ out) {
  const int row = blockIdx.x;
  const int tid = threadIdx.x;
  __shared__ float s1[256], s2[256];
  const float* xr = x + (size_t)row * DM;
  float v[6]; float sum = 0.f, sq = 0.f;
#pragma unroll
  for (int i = 0; i < 6; ++i) { v[i] = xr[tid + 256 * i]; sum += v[i]; sq += v[i] * v[i]; }
  s1[tid] = sum; s2[tid] = sq; __syncthreads();
  for (int o = 128; o > 0; o >>= 1) {
    if (tid < o) { s1[tid] += s1[tid + o]; s2[tid] += s2[tid + o]; }
    __syncthreads();
  }
  const float mean = s1[0] * (1.0f / DM);
  const float var  = s2[0] * (1.0f / DM) - mean * mean;
  const float rinv = rsqrtf(var + EPSF);
  float* orow = out + (size_t)row * DM;
#pragma unroll
  for (int i = 0; i < 6; ++i) {
    const int d = tid + 256 * i;
    float y = (v[i] - mean) * rinv;
    if (g) y = y * g[d] + b[d];
    if (e6) {
      const float sft = e6[((size_t)row * 6 + ish) * DM + d];
      const float scl = e6[((size_t)row * 6 + isc) * DM + d];
      y = y * (1.0f + scl) + sft;
    }
    orow[d] = y;
  }
}

// ---------------- RMSNorm (in place): x = x * rsqrt(mean(x^2)+eps) * g ----------------
__global__ __launch_bounds__(256)
void rms_kernel(float* __restrict__ x, const float* __restrict__ g) {
  const int row = blockIdx.x;
  const int tid = threadIdx.x;
  __shared__ float s1[256];
  float* xr = x + (size_t)row * DM;
  float v[6]; float sq = 0.f;
#pragma unroll
  for (int i = 0; i < 6; ++i) { v[i] = xr[tid + 256 * i]; sq += v[i] * v[i]; }
  s1[tid] = sq; __syncthreads();
  for (int o = 128; o > 0; o >>= 1) {
    if (tid < o) s1[tid] += s1[tid + o];
    __syncthreads();
  }
  const float rinv = rsqrtf(s1[0] * (1.0f / DM) + EPSF);
#pragma unroll
  for (int i = 0; i < 6; ++i) {
    const int d = tid + 256 * i;
    xr[d] = v[i] * rinv * g[d];
  }
}

// ---------------- RoPE over [S, H, 64, 2] interleaved pairs ----------------
__global__ void rope_kernel(const float* __restrict__ in, const float* __restrict__ cs,
                            const float* __restrict__ sn, float* __restrict__ out, int total) {
  int idx = blockIdx.x * 256 + threadIdx.x;
  if (idx >= total) return;
  const int i = idx & 63;
  const int h = (idx >> 6) % NH;
  const int s = idx / (64 * NH);
  const float c  = cs[(size_t)s * 64 + i];
  const float si = sn[(size_t)s * 64 + i];
  const size_t base = (size_t)s * DM + h * HDM + 2 * i;
  const float x1 = in[base], x2 = in[base + 1];
  out[base]     = x1 * c - x2 * si;
  out[base + 1] = x1 * si + x2 * c;
}

// ---------------- softmax over rows of [rows][L], with pre-scale ----------------
__global__ __launch_bounds__(256)
void softmax_kernel(float* __restrict__ p, int L) {
  const int row = blockIdx.x;
  const int tid = threadIdx.x;
  float* r = p + (size_t)row * L;
  __shared__ float red[256];
  float mx = -3.4e38f;
  for (int i = tid; i < L; i += 256) mx = fmaxf(mx, r[i] * ATT_SCALE);
  red[tid] = mx; __syncthreads();
  for (int o = 128; o > 0; o >>= 1) {
    if (tid < o) red[tid] = fmaxf(red[tid], red[tid + o]);
    __syncthreads();
  }
  mx = red[0]; __syncthreads();
  float sum = 0.f;
  for (int i = tid; i < L; i += 256) { float e = expf(r[i] * ATT_SCALE - mx); r[i] = e; sum += e; }
  red[tid] = sum; __syncthreads();
  for (int o = 128; o > 0; o >>= 1) {
    if (tid < o) red[tid] += red[tid + o];
    __syncthreads();
  }
  const float inv = 1.0f / red[0];
  for (int i = tid; i < L; i += 256) r[i] *= inv;
}

// ---------------- out = x + y * (e6 ? e6[s,idx,:] : 1) ----------------
__global__ void axpy_mod_kernel(const float* __restrict__ x, const float* __restrict__ y,
                                const float* __restrict__ e6, int idx, float* __restrict__ out, size_t n) {
  size_t i = (size_t)blockIdx.x * 256 + threadIdx.x;
  if (i >= n) return;
  float m = 1.0f;
  if (e6) {
    const size_t s = i / DM;
    const int d = (int)(i % DM);
    m = e6[(s * 6 + idx) * DM + d];
  }
  out[i] = x[i] + y[i] * m;
}

// =====================================================================================
extern "C" void kernel_launch(void* const* d_in, const int* in_sizes, int n_in,
                              void* d_out, int out_size, void* d_ws, size_t ws_size,
                              hipStream_t stream) {
  (void)in_sizes; (void)n_in; (void)out_size; (void)ws_size;

  // inputs in setup_inputs() insertion order
  const float* vid_in = (const float*)d_in[0];
  const float* aud_in = (const float*)d_in[1];
  const float* vid_e  = (const float*)d_in[2];
  const float* aud_e  = (const float*)d_in[3];
  const float* vctx   = (const float*)d_in[4];
  const float* actx   = (const float*)d_in[5];
  const float* vcos   = (const float*)d_in[6];
  const float* vsin   = (const float*)d_in[7];
  const float* acos_  = (const float*)d_in[8];
  const float* asin_  = (const float*)d_in[9];
  // params: mod_w,mod_b, sq_w,sq_b, sk_w,sk_b, sv_w,sv_b, so_w,so_b, s_nq,s_nk,
  //         cq_w,cq_b, ck_w,ck_b, cv_w,cv_b, co_w,co_b, c_nq,c_nk,
  //         kf_w,kf_b, vf_w,vf_b, pre_g,pre_b, nkf, n3_g,n3_b, f1_w,f1_b, f2_w,f2_b  (35)
  const float* VP[35]; const float* AP[35];
  for (int i = 0; i < 35; ++i) { VP[i] = (const float*)d_in[10 + i]; AP[i] = (const float*)d_in[45 + i]; }

  char* wsb = (char*)d_ws; size_t off = 0;
  auto alloc = [&](size_t bytes) -> void* {
    void* p = wsb + off; off = (off + bytes + 255) & ~(size_t)255; return p;
  };

  struct WB { __bf16 *mod,*sq,*sk,*sv,*so,*cq,*ck,*cv,*co,*kf,*vf,*f1,*f2; };
  auto allocWB = [&](WB& w) {
    const size_t dd = (size_t)DM * DM * sizeof(__bf16), df = (size_t)DM * FF * sizeof(__bf16);
    w.mod=(__bf16*)alloc(dd); w.sq=(__bf16*)alloc(dd); w.sk=(__bf16*)alloc(dd); w.sv=(__bf16*)alloc(dd);
    w.so=(__bf16*)alloc(dd); w.cq=(__bf16*)alloc(dd); w.ck=(__bf16*)alloc(dd); w.cv=(__bf16*)alloc(dd);
    w.co=(__bf16*)alloc(dd); w.kf=(__bf16*)alloc(dd); w.vf=(__bf16*)alloc(dd);
    w.f1=(__bf16*)alloc(df); w.f2=(__bf16*)alloc(df);
  };
  WB WV, WA; allocWB(WV); allocWB(WA);

  float*  e6_v = (float*)alloc((size_t)SV * 6 * DM * 4);
  float*  e6_a = (float*)alloc((size_t)SA * 6 * DM * 4);
  float*  xv   = (float*)alloc((size_t)SV * DM * 4);
  float*  xa   = (float*)alloc((size_t)SA * DM * 4);
  float*  og   = (float*)alloc((size_t)SA * DM * 4);
  float*  t0   = (float*)alloc((size_t)SV * DM * 4);
  float*  qb   = (float*)alloc((size_t)SV * DM * 4);
  float*  kb   = (float*)alloc((size_t)SV * DM * 4);
  float*  vb   = (float*)alloc((size_t)SV * DM * 4);
  float*  qr   = (float*)alloc((size_t)SV * DM * 4);
  float*  ao   = (float*)alloc((size_t)SV * DM * 4);
  float*  ao2  = (float*)alloc((size_t)SV * DM * 4);
  float*  yb   = (float*)alloc((size_t)SV * DM * 4);
  float*  hb   = (float*)alloc((size_t)SV * FF * 4);
  float*  sc   = (float*)alloc((size_t)SV * SV * 4);
  __bf16* kbf  = (__bf16*)alloc((size_t)SV * DM * 2);
  __bf16* vt   = (__bf16*)alloc((size_t)DM * SV * 2);

  auto gemm = [&](const float* A, int lda, const __bf16* Bt, int ldb, const float* bias,
                  float* C, int ldc, int M, int N, int K, int epi) {
    dim3 g(N / BN, M / BM, 1), blk(256, 1, 1);
    if (epi == 0) gemm_bf16_kernel<0><<<g, blk, 0, stream>>>(A, lda, Bt, ldb, bias, C, ldc, K);
    else          gemm_bf16_kernel<1><<<g, blk, 0, stream>>>(A, lda, Bt, ldb, bias, C, ldc, K);
  };
  auto convW = [&](const float* w, __bf16* dst, int R, int Cc) {
    dim3 g(Cc / 32, R / 32, 1), blk(32, 8, 1);
    transpose_conv_kernel<<<g, blk, 0, stream>>>(w, dst, R, Cc);
  };
  auto convAll = [&](const float* const* P, WB& w) {
    convW(P[0],  w.mod, DM, DM);
    convW(P[2],  w.sq,  DM, DM); convW(P[4],  w.sk, DM, DM);
    convW(P[6],  w.sv,  DM, DM); convW(P[8],  w.so, DM, DM);
    convW(P[12], w.cq,  DM, DM); convW(P[14], w.ck, DM, DM);
    convW(P[16], w.cv,  DM, DM); convW(P[18], w.co, DM, DM);
    convW(P[22], w.kf,  DM, DM); convW(P[24], w.vf, DM, DM);
    convW(P[31], w.f1,  DM, FF); convW(P[33], w.f2, FF, DM);
  };
  auto ln = [&](const float* x, const float* g, const float* b, const float* e6,
                int ish, int isc, float* out_, int S) {
    ln_kernel<<<S, 256, 0, stream>>>(x, g, b, e6, ish, isc, out_);
  };
  auto rms = [&](float* x, const float* g, int S) { rms_kernel<<<S, 256, 0, stream>>>(x, g); };
  auto rope = [&](const float* in, const float* cs, const float* sn, float* out_, int S) {
    const int total = S * NH * 64;
    rope_kernel<<<(total + 255) / 256, 256, 0, stream>>>(in, cs, sn, out_, total);
  };
  auto axpy = [&](const float* x, const float* y, const float* e6, int idx, float* out_, int S) {
    const size_t n = (size_t)S * DM;
    axpy_mod_kernel<<<(int)((n + 255) / 256), 256, 0, stream>>>(x, y, e6, idx, out_, n);
  };
  // attention: q [Sq,DM] f32, k/v [Sk,DM] f32 -> out [Sq,DM] f32 (per-head WMMA GEMMs)
  auto attn = [&](const float* q, const float* k, const float* v, int Sq, int Sk, float* out_) {
    const size_t nk = (size_t)Sk * DM;
    conv_bf16_kernel<<<(int)((nk + 255) / 256), 256, 0, stream>>>(k, kbf, nk);
    {
      dim3 g(DM / 32, Sk / 32, 1), blk(32, 8, 1);
      transpose_conv_kernel<<<g, blk, 0, stream>>>(v, vt, Sk, DM);   // -> vt [DM][Sk]
    }
    for (int h = 0; h < NH; ++h) {
      gemm(q + h * HDM, DM, kbf + h * HDM, DM, nullptr, sc, Sk, Sq, Sk, HDM, 0);
      softmax_kernel<<<Sq, 256, 0, stream>>>(sc, Sk);
      gemm(sc, Sk, vt + (size_t)h * HDM * Sk, Sk, nullptr, out_ + h * HDM, DM, Sq, HDM, Sk, 0);
    }
  };

  auto modality_first = [&](float* xbuf, const float* eraw, float* e6, const float* const* P, WB& W,
                            const float* cs, const float* sn, int S) {
    gemm(eraw, DM, W.mod, DM, P[1], e6, DM, 6 * S, DM, DM, 0);        // e6 = e_raw @ mod + b
    ln(xbuf, nullptr, nullptr, e6, 0, 1, t0, S);                      // ln(x)*(1+e1)+e0
    gemm(t0, DM, W.sq, DM, P[3], qb, DM, S, DM, DM, 0);
    gemm(t0, DM, W.sk, DM, P[5], kb, DM, S, DM, DM, 0);
    gemm(t0, DM, W.sv, DM, P[7], vb, DM, S, DM, DM, 0);
    rms(qb, P[10], S); rms(kb, P[11], S);
    rope(qb, cs, sn, qb, S); rope(kb, cs, sn, kb, S);
    attn(qb, kb, vb, S, S, ao);
    gemm(ao, DM, W.so, DM, P[9], yb, DM, S, DM, DM, 0);
    axpy(xbuf, yb, e6, 2, xbuf, S);                                   // x += y*e2
  };

  auto cross_ffn = [&](float* xbuf, float* e6, const float* ctx, const float* tgt,
                       const float* const* P, WB& W,
                       const float* cs_s, const float* sn_s, const float* cs_t, const float* sn_t,
                       int S, int St) {
    ln(xbuf, P[29], P[30], nullptr, 0, 0, t0, S);                     // xn = ln(x, n3)
    gemm(t0, DM, W.cq, DM, P[13], qb, DM, S, DM, DM, 0);  rms(qb, P[20], S);
    gemm(ctx, DM, W.ck, DM, P[15], kb, DM, CTXN, DM, DM, 0); rms(kb, P[21], CTXN);
    gemm(ctx, DM, W.cv, DM, P[17], vb, DM, CTXN, DM, DM, 0);
    attn(qb, kb, vb, S, CTXN, ao);                                    // xa = attn(q,k,v)
    ln(tgt, P[26], P[27], nullptr, 0, 0, t0, St);                     // t = ln(tgt, pre)
    gemm(t0, DM, W.kf, DM, P[23], kb, DM, St, DM, DM, 0); rms(kb, P[28], St);
    gemm(t0, DM, W.vf, DM, P[25], vb, DM, St, DM, DM, 0);
    rope(qb, cs_s, sn_s, qr, S);                                      // rope(q)
    rope(kb, cs_t, sn_t, kb, St);                                     // rope(kt)
    attn(qr, kb, vb, S, St, ao2);
    axpy(ao, ao2, nullptr, 0, ao, S);                                 // xa += attn2
    gemm(ao, DM, W.co, DM, P[19], yb, DM, S, DM, DM, 0);
    axpy(xbuf, yb, nullptr, 0, xbuf, S);                              // x += co(xa)
    ln(xbuf, nullptr, nullptr, e6, 3, 4, t0, S);                      // ln(x)*(1+e4)+e3
    gemm(t0, DM, W.f1, DM, P[32], hb, FF, S, FF, DM, 1);              // gelu epilogue
    gemm(hb, FF, W.f2, FF, P[34], yb, DM, S, DM, FF, 0);
    axpy(xbuf, yb, e6, 5, xbuf, S);                                   // x += y*e5
  };

  // -------- pipeline (matches reference _fusion_block order) --------
  convAll(VP, WV);
  convAll(AP, WA);
  (void)hipMemcpyAsync(xa, aud_in, (size_t)SA * DM * 4, hipMemcpyDeviceToDevice, stream);
  (void)hipMemcpyAsync(xv, vid_in, (size_t)SV * DM * 4, hipMemcpyDeviceToDevice, stream);

  modality_first(xa, aud_e, e6_a, AP, WA, acos_, asin_, SA);
  modality_first(xv, vid_e, e6_v, VP, WV, vcos, vsin, SV);
  (void)hipMemcpyAsync(og, xa, (size_t)SA * DM * 4, hipMemcpyDeviceToDevice, stream);
  cross_ffn(xa, e6_a, actx, xv, AP, WA, acos_, asin_, vcos, vsin, SA, SV);
  cross_ffn(xv, e6_v, vctx, og, VP, WV, vcos, vsin, acos_, asin_, SV, SA);

  // outputs: (vid, audio) concatenated flat
  (void)hipMemcpyAsync(d_out, xv, (size_t)SV * DM * 4, hipMemcpyDeviceToDevice, stream);
  (void)hipMemcpyAsync((float*)d_out + (size_t)SV * DM, xa, (size_t)SA * DM * 4,
                       hipMemcpyDeviceToDevice, stream);
}
